// AttentionGuidedTextureDiffusion_24592982736956
// MI455X (gfx1250) — matrix operand order repaired and checked by reading, not compiled, not run on hardware
//
#include <hip/hip_runtime.h>

typedef _Float16 v8h  __attribute__((ext_vector_type(8)));
typedef _Float16 v16h __attribute__((ext_vector_type(16)));
typedef float    v8f  __attribute__((ext_vector_type(8)));

constexpr int Bb = 2;
constexpr int Hh = 192;
constexpr int Ww = 192;
constexpr int HW = Hh * Ww;          // 36864
constexpr int PIT = Ww + 2;          // 194 (zero-pad ring of 1)
constexpr int PADPIX = PIT * PIT;    // 37636
constexpr int LAT = 24;
constexpr int RR = 7;
constexpr int TAPS = 49;
constexpr int COUT2 = 1216;          // 1176 padded up to 76 tiles of 16 (divisible by NT=4)

// ---------------------------------------------------------------- zero fill
__global__ void zero_kernel(uint4* p, long long n) {
  long long i = (long long)blockIdx.x * blockDim.x + threadIdx.x;
  long long stride = (long long)gridDim.x * blockDim.x;
  uint4 z; z.x = 0u; z.y = 0u; z.z = 0u; z.w = 0u;
  for (; i < n; i += stride) p[i] = z;
}

// ------------------------------------------------ weight repack to B-fragment
// wp[((chunk*coTiles + ct)*32 + lane)*16 + e]
// chunk = tap*(CIN/32) + cc ; ci = cc*32 + (e>>3)*16 + (lane>>4)*8 + (e&7) ; co = ct*16 + (lane&15)
__global__ void repack_kernel(const float* __restrict__ w, _Float16* __restrict__ wp,
                              int CIN, int coTiles, int coReal, int KS, long long total) {
  long long idx = (long long)blockIdx.x * blockDim.x + threadIdx.x;
  if (idx >= total) return;
  int e    = (int)(idx & 15);
  int lane = (int)((idx >> 4) & 31);
  long long rest = idx >> 9;
  int ct    = (int)(rest % coTiles);
  int chunk = (int)(rest / coTiles);
  int cch = CIN >> 5;
  int tap = chunk / cch;
  int ci  = (chunk % cch) * 32 + (e >> 3) * 16 + (lane >> 4) * 8 + (e & 7);
  int co  = ct * 16 + (lane & 15);
  float v = 0.f;
  if (co < coReal) v = w[(long long)(co * CIN + ci) * (KS * KS) + tap];
  wp[idx] = (_Float16)v;
}

// ---------------------------------------------- WMMA implicit-GEMM conv
// One wave computes an (MT*16 pixel) x (NT*16 cout) super-tile: A fragments are
// reused across NT cout tiles, B fragments across MT pixel tiles.
// Input: f16 NHWC (interior pointer of a zero-ring-padded buffer for KS=3).
template<int CIN, int KS, int MT, int NT, bool RELU, bool PLANAR>
__global__ __launch_bounds__(32)
void conv_wmma_kernel(const _Float16* __restrict__ in, int inPitch, int inStride, long long inBatch,
                      const _Float16* __restrict__ wp, const float* __restrict__ bias,
                      _Float16* __restrict__ out, int outPitch, int outStride, long long outBatch,
                      int coTiles, int coReal) {
  const int lane = threadIdx.x;             // 0..31, no divergence: EXEC all ones
  const int tileX = blockIdx.x;
  const int ctBase = blockIdx.y * NT;       // first cout tile of this wave
  constexpr int TPR = Ww / (16 * MT);       // super-tiles per row
  const int x0 = (tileX % TPR) * (16 * MT);
  const int y  = (tileX / TPR) % Hh;
  const int b  = tileX / (TPR * Hh);
  const int mrow = lane & 15;               // A-matrix row (pixel within tile)
  const int hs   = lane >> 4;               // half-select for K packing
  const _Float16* ib = in + (long long)b * inBatch;
  const v16h* wv = (const v16h*)wp;

  v8f acc[MT][NT];
#pragma unroll
  for (int nt = 0; nt < NT; ++nt) {
    const int nG = (ctBase + nt) * 16 + (lane & 15);
    const float bval = (nG < coReal) ? bias[nG] : 0.f;
#pragma unroll
    for (int mt = 0; mt < MT; ++mt)
#pragma unroll
      for (int r = 0; r < 8; ++r) acc[mt][nt][r] = bval;
  }

  constexpr int CCH = CIN / 32;
#pragma unroll
  for (int t = 0; t < KS * KS; ++t) {
    const int dy = t / KS - KS / 2;
    const int dx = t % KS - KS / 2;
#pragma unroll
    for (int cc = 0; cc < CCH; ++cc) {
      const int c0 = cc * 32;
      // A fragments: lane<16 holds K = 0..7 & 16..23 ; lane>=16 holds K = 8..15 & 24..31
      v16h a[MT];
#pragma unroll
      for (int mt = 0; mt < MT; ++mt) {
        const int xa = x0 + mt * 16 + mrow;
        const _Float16* prow =
            ib + ((long long)(y + dy) * inPitch + (xa + dx)) * inStride + hs * 8 + c0;
        v8h lo = *(const v8h*)(prow);
        v8h hi = *(const v8h*)(prow + 16);
#pragma unroll
        for (int e = 0; e < 8; ++e) { a[mt][e] = lo[e]; a[mt][e + 8] = hi[e]; }
      }
      const int chunk = t * CCH + cc;
#pragma unroll
      for (int nt = 0; nt < NT; ++nt) {
        v16h bf = wv[(long long)(chunk * coTiles + (ctBase + nt)) * 32 + lane];
#pragma unroll
        for (int mt = 0; mt < MT; ++mt) {
          acc[mt][nt] = __builtin_amdgcn_wmma_f32_16x16x32_f16(false, a[mt], false, bf,
                                                              (short)0, acc[mt][nt],
                                                              false, false);
        }
      }
    }
  }

#pragma unroll
  for (int mt = 0; mt < MT; ++mt) {
#pragma unroll
    for (int nt = 0; nt < NT; ++nt) {
      v8f r8 = acc[mt][nt];
      if (RELU) {
#pragma unroll
        for (int r = 0; r < 8; ++r) r8[r] = fmaxf(r8[r], 0.f);
      }
      const int nG = (ctBase + nt) * 16 + (lane & 15);
      if (PLANAR) {
        // D layout: lane holds column n, rows m = 8*hs + r -> consecutive pixels
        v8h o;
#pragma unroll
        for (int r = 0; r < 8; ++r) o[r] = (_Float16)r8[r];
        const long long pix0 = (long long)y * Ww + x0 + mt * 16 + hs * 8;
        *(v8h*)(out + (long long)b * outBatch + (long long)nG * HW + pix0) = o;
      } else {
#pragma unroll
        for (int r = 0; r < 8; ++r) {
          const int xx = x0 + mt * 16 + hs * 8 + r;
          out[(long long)b * outBatch +
              ((long long)y * outPitch + xx) * outStride + nG] = (_Float16)r8[r];
        }
      }
    }
  }
}

// ------------------------------------- small direct convs (3->64, 1->32) + ReLU
template<int CIN, int COUT>
__global__ void conv_direct_kernel(const float* __restrict__ x, const float* __restrict__ w,
                                   const float* __restrict__ bias, _Float16* __restrict__ out,
                                   long long outBatch) {
  int idx = blockIdx.x * blockDim.x + threadIdx.x;
  if (idx >= Bb * HW) return;
  const int b = idx / HW;
  const int pix = idx - b * HW;
  const int y = pix / Ww, xx = pix - y * Ww;
  float patch[CIN * 9];
#pragma unroll
  for (int ci = 0; ci < CIN; ++ci) {
#pragma unroll
    for (int t = 0; t < 9; ++t) {
      const int dy = t / 3 - 1, dx = t % 3 - 1;
      const int yy = y + dy, xc = xx + dx;
      float v = 0.f;
      if (yy >= 0 && yy < Hh && xc >= 0 && xc < Ww)
        v = x[(long long)(b * CIN + ci) * HW + yy * Ww + xc];
      patch[ci * 9 + t] = v;
    }
  }
  _Float16* op = out + (long long)b * outBatch + ((long long)y * PIT + xx) * COUT;
#pragma unroll 4
  for (int co = 0; co < COUT; ++co) {
    float acc = bias[co];
#pragma unroll
    for (int ci = 0; ci < CIN; ++ci)
#pragma unroll
      for (int t = 0; t < 9; ++t)
        acc += w[(co * CIN + ci) * 9 + t] * patch[ci * 9 + t];
    op[co] = (_Float16)fmaxf(acc, 0.f);
  }
}

// ---------------- gate = sigm(sg2(sg1)) * (1 + 0.5*sigm(ue2(ue1))*umap); attw = gate*attmap
__global__ void gate_kernel(const _Float16* __restrict__ sg1, const _Float16* __restrict__ ue1,
                            const float* __restrict__ sgw, const float* __restrict__ sgb,
                            const float* __restrict__ uew, const float* __restrict__ ueb,
                            const float* __restrict__ attmap, const float* __restrict__ umap,
                            float* __restrict__ gate, float* __restrict__ attw) {
  int idx = blockIdx.x * blockDim.x + threadIdx.x;
  if (idx >= Bb * HW) return;
  const _Float16* s = sg1 + (long long)idx * 32;
  float g = sgb[0];
#pragma unroll
  for (int i = 0; i < 32; ++i) g += (float)s[i] * sgw[i];
  const _Float16* u = ue1 + (long long)idx * 64;
  float uu = ueb[0];
#pragma unroll
  for (int i = 0; i < 64; ++i) uu += (float)u[i] * uew[i];
  g  = 1.f / (1.f + __expf(-g));
  uu = 1.f / (1.f + __expf(-uu));
  const float gg = g * (1.f + 0.5f * uu * umap[idx]);
  gate[idx] = gg;
  attw[idx] = gg * attmap[idx];
}

// ---------------- in-place softmax over 49 taps (logits scaled by gate first)
__global__ void softmax_kernel(_Float16* __restrict__ wbuf, const float* __restrict__ gate) {
  long long idx = (long long)blockIdx.x * blockDim.x + threadIdx.x;
  if (idx >= (long long)Bb * LAT * HW) return;
  const int pix = (int)(idx % HW);
  const int c   = (int)((idx / HW) % LAT);
  const int b   = (int)(idx / ((long long)LAT * HW));
  const float g = gate[b * HW + pix];
  _Float16* base = wbuf + (long long)(b * COUT2 + c * TAPS) * HW + pix;
  float v[TAPS];
  float mx = -3.0e38f;
#pragma unroll
  for (int k = 0; k < TAPS; ++k) {
    v[k] = (float)base[(long long)k * HW] * g;
    mx = fmaxf(mx, v[k]);
  }
  float s = 0.f;
#pragma unroll
  for (int k = 0; k < TAPS; ++k) { v[k] = __expf(v[k] - mx); s += v[k]; }
  const float inv = 1.f / s;
#pragma unroll
  for (int k = 0; k < TAPS; ++k) base[(long long)k * HW] = (_Float16)(v[k] * inv);
}

// ---------------- one diffusion step: 49-tap dynamic filter, edge padding via clamp
__global__ void diffusion_kernel(const float* __restrict__ src, float* __restrict__ dst,
                                 const _Float16* __restrict__ wbuf) {
  long long idx = (long long)blockIdx.x * blockDim.x + threadIdx.x;
  if (idx >= (long long)Bb * LAT * HW) return;
  const int pix = (int)(idx % HW);
  const int c   = (int)((idx / HW) % LAT);
  const int b   = (int)(idx / ((long long)LAT * HW));
  const int y = pix / Ww, x = pix - y * Ww;
  const _Float16* wb = wbuf + (long long)(b * COUT2 + c * TAPS) * HW + pix;
  const float* sp = src + (long long)(b * LAT + c) * HW;
  float acc = 0.f;
#pragma unroll
  for (int i = 0; i < RR; ++i) {
    const int yy = min(max(y + i - 3, 0), Hh - 1);
    const float* row = sp + yy * Ww;
#pragma unroll
    for (int j = 0; j < RR; ++j) {
      const int xc = min(max(x + j - 3, 0), Ww - 1);
      acc += (float)wb[(long long)(i * RR + j) * HW] * row[xc];
    }
  }
  dst[(long long)(b * LAT + c) * HW + pix] = acc;
}

// =============================================================== host launch
extern "C" void kernel_launch(void* const* d_in, const int* in_sizes, int n_in,
                              void* d_out, int out_size, void* d_ws, size_t ws_size,
                              hipStream_t stream) {
  (void)in_sizes; (void)n_in; (void)out_size; (void)ws_size;
  const float* depth  = (const float*)d_in[0];
  const float* texf   = (const float*)d_in[1];
  const float* attmap = (const float*)d_in[2];
  const float* umap   = (const float*)d_in[3];
  const float* te_w1 = (const float*)d_in[4];  const float* te_b1 = (const float*)d_in[5];
  const float* te_w2 = (const float*)d_in[6];  const float* te_b2 = (const float*)d_in[7];
  const float* ae_w1 = (const float*)d_in[8];  const float* ae_b1 = (const float*)d_in[9];
  const float* ae_w2 = (const float*)d_in[10]; const float* ae_b2 = (const float*)d_in[11];
  const float* fu_w1 = (const float*)d_in[12]; const float* fu_b1 = (const float*)d_in[13];
  const float* fu_w2 = (const float*)d_in[14]; const float* fu_b2 = (const float*)d_in[15];
  const float* sg_w1 = (const float*)d_in[16]; const float* sg_b1 = (const float*)d_in[17];
  const float* sg_w2 = (const float*)d_in[18]; const float* sg_b2 = (const float*)d_in[19];
  const float* ue_w1 = (const float*)d_in[20]; const float* ue_b1 = (const float*)d_in[21];
  const float* ue_w2 = (const float*)d_in[22]; const float* ue_b2 = (const float*)d_in[23];

  float* out_enh  = (float*)d_out;
  float* out_gate = out_enh + (long long)Bb * LAT * HW;
  float* out_attw = out_gate + (long long)Bb * HW;

  char* ws = (char*)d_ws;
  size_t off = 0;
  auto alloc = [&](size_t bytes) -> void* {
    void* p = ws + off;
    off += (bytes + 255) & ~(size_t)255;
    return p;
  };
  // padded NHWC f16 activation buffers first (one contiguous zero region)
  _Float16* te1buf = (_Float16*)alloc((size_t)Bb * PADPIX * 64 * 2);
  _Float16* ae1buf = (_Float16*)alloc((size_t)Bb * PADPIX * 32 * 2);
  _Float16* fusbuf = (_Float16*)alloc((size_t)Bb * PADPIX * 192 * 2);   // tex(0..127)|att(128..191)
  const size_t zeroBytes = off;
  _Float16* sg1buf = (_Float16*)alloc((size_t)Bb * HW * 32 * 2);
  _Float16* ue1buf = (_Float16*)alloc((size_t)Bb * HW * 64 * 2);
  _Float16* fu1buf = (_Float16*)alloc((size_t)Bb * HW * 256 * 2);
  _Float16* fu2buf = (_Float16*)alloc((size_t)Bb * HW * COUT2 * 2);     // planar [b][co][pix], ~L2-resident
  float*    latA   = (float*)alloc((size_t)Bb * LAT * HW * 4);
  auto wpBytes = [](int CIN, int coTiles, int KS) {
    return (size_t)(KS * KS * CIN / 32) * coTiles * 512 * 2;
  };
  _Float16* wp_te2 = (_Float16*)alloc(wpBytes(64, 8, 3));
  _Float16* wp_ae2 = (_Float16*)alloc(wpBytes(32, 4, 3));
  _Float16* wp_sg1 = (_Float16*)alloc(wpBytes(64, 2, 3));
  _Float16* wp_ue1 = (_Float16*)alloc(wpBytes(128, 4, 3));
  _Float16* wp_fu1 = (_Float16*)alloc(wpBytes(192, 16, 3));
  _Float16* wp_fu2 = (_Float16*)alloc(wpBytes(256, 76, 1));

  // 1) zero the padded buffers (rings must be 0 for the 3x3 WMMA convs)
  zero_kernel<<<2048, 256, 0, stream>>>((uint4*)ws, (long long)(zeroBytes / 16));

  // 2) repack weights into WMMA B-fragment order (f32 -> f16)
  auto repack = [&](const float* w, _Float16* wp, int CIN, int coTiles, int coReal, int KS) {
    long long total = (long long)(KS * KS * CIN / 32) * coTiles * 512;
    repack_kernel<<<(int)((total + 255) / 256), 256, 0, stream>>>(w, wp, CIN, coTiles, coReal, KS, total);
  };
  repack(te_w2, wp_te2, 64, 8, 128, 3);
  repack(ae_w2, wp_ae2, 32, 4, 64, 3);
  repack(sg_w1, wp_sg1, 64, 2, 32, 3);
  repack(ue_w1, wp_ue1, 128, 4, 64, 3);
  repack(fu_w1, wp_fu1, 192, 16, 256, 3);
  repack(fu_w2, wp_fu2, 256, 76, 1176, 1);

  _Float16* te1int = te1buf + (size_t)(PIT + 1) * 64;
  _Float16* ae1int = ae1buf + (size_t)(PIT + 1) * 32;
  _Float16* fusint = fusbuf + (size_t)(PIT + 1) * 192;

  // 3) input encoders (tiny K: direct VALU), fp32 NCHW -> f16 NHWC padded interior
  {
    const int n = Bb * HW;
    conv_direct_kernel<3, 64><<<(n + 127) / 128, 128, 0, stream>>>(texf, te_w1, te_b1, te1int, (long long)PADPIX * 64);
    conv_direct_kernel<1, 32><<<(n + 127) / 128, 128, 0, stream>>>(attmap, ae_w1, ae_b1, ae1int, (long long)PADPIX * 32);
  }

  // 4) WMMA convs: MT=2 pixel tiles x NT cout tiles per wave (register blocking)
  const dim3 blk(32, 1, 1);
  const int superTiles = Bb * Hh * (Ww / 32);   // MT=2 -> 2304, exact coverage, no divergence
  conv_wmma_kernel<64, 3, 2, 4, true, false><<<dim3(superTiles, 2), blk, 0, stream>>>(
      te1int, PIT, 64, (long long)PADPIX * 64, wp_te2, te_b2,
      fusint, PIT, 192, (long long)PADPIX * 192, 8, 128);                    // te2 -> fused ch0..127
  conv_wmma_kernel<32, 3, 2, 4, true, false><<<dim3(superTiles, 1), blk, 0, stream>>>(
      ae1int, PIT, 32, (long long)PADPIX * 32, wp_ae2, ae_b2,
      fusint + 128, PIT, 192, (long long)PADPIX * 192, 4, 64);               // ae2 -> fused ch128..191
  conv_wmma_kernel<128, 3, 2, 4, true, false><<<dim3(superTiles, 1), blk, 0, stream>>>(
      fusint, PIT, 192, (long long)PADPIX * 192, wp_ue1, ue_b1,
      ue1buf, Ww, 64, (long long)HW * 64, 4, 64);                            // ue1(tex)
  conv_wmma_kernel<64, 3, 2, 2, true, false><<<dim3(superTiles, 1), blk, 0, stream>>>(
      fusint + 128, PIT, 192, (long long)PADPIX * 192, wp_sg1, sg_b1,
      sg1buf, Ww, 32, (long long)HW * 32, 2, 32);                            // sg1(att)
  conv_wmma_kernel<192, 3, 2, 4, true, false><<<dim3(superTiles, 4), blk, 0, stream>>>(
      fusint, PIT, 192, (long long)PADPIX * 192, wp_fu1, fu_b1,
      fu1buf, Ww, 256, (long long)HW * 256, 16, 256);                        // fu1(concat)
  conv_wmma_kernel<256, 1, 2, 4, false, true><<<dim3(superTiles, 19), blk, 0, stream>>>(
      fu1buf, Ww, 256, (long long)HW * 256, wp_fu2, fu_b2,
      fu2buf, Ww, 0, (long long)HW * COUT2, 76, 1176);                       // fu2 1x1, planar out

  // 5) gate + attention_weights (written straight into d_out)
  {
    const int n = Bb * HW;
    gate_kernel<<<(n + 255) / 256, 256, 0, stream>>>(sg1buf, ue1buf, sg_w2, sg_b2,
                                                     ue_w2, ue_b2, attmap, umap,
                                                     out_gate, out_attw);
  }

  // 6) gated softmax over 49 taps, in place on the planar f16 buffer
  {
    const long long n = (long long)Bb * LAT * HW;
    softmax_kernel<<<(int)((n + 255) / 256), 256, 0, stream>>>(fu2buf, out_gate);
  }

  // 7) 16 diffusion steps; ping-pong latA <-> d_out so step 16 lands in d_out
  {
    const long long n = (long long)Bb * LAT * HW;
    const int g = (int)((n + 255) / 256);
    for (int s = 0; s < 16; ++s) {
      const float* src = (s == 0) ? depth : ((s & 1) ? latA : out_enh);
      float* dst = (s & 1) ? out_enh : latA;
      diffusion_kernel<<<g, 256, 0, stream>>>(src, dst, fu2buf);
    }
  }
}
